// CircularSplineLayer_34574486733367
// MI455X (gfx1250) — matrix operand order repaired and checked by reading, not compiled, not run on hardware
//
#include <hip/hip_runtime.h>
#include <hip/hip_bf16.h>
#include <math.h>

typedef __attribute__((ext_vector_type(16))) __bf16 v16bf;
typedef __attribute__((ext_vector_type(8)))  float  v8f;

#define B_N     1024
#define S_IN    4096
#define S_OUT   4096
#define KBINS   8
#define HID     512
#define NCH     24              /* 3*K */
#define W2COLS  98304           /* S_OUT*NCH */
#define NPAIRS  2048            /* S_OUT/2  */
#define TWO_PI_F 6.2831853071795864769f
#define EPS_F    1e-6f

union V16 { uint4 q[2]; v16bf v; };   // 32B: one bf16 WMMA fragment

__device__ __forceinline__ float fast_rcp(float x) { return __builtin_amdgcn_rcpf(x); }

__device__ __forceinline__ float fast_tanh(float x) {
    const float t = __expf(-2.0f * fabsf(x));
    const float r = (1.0f - t) * fast_rcp(1.0f + t);
    return copysignf(r, x);
}

// ---------------- stage 1: deterministic mean/var reduction ----------------
__global__ __launch_bounds__(256) void k_reduce1(const float* __restrict__ x,
                                                 float* __restrict__ p_sum,
                                                 float* __restrict__ p_sq) {
    __shared__ float ss[256], sq[256];
    const int t = threadIdx.x;
    const size_t base = (size_t)blockIdx.x * 2048;
    float s = 0.f, q = 0.f;
    for (int i = t; i < 2048; i += 256) { float v = x[base + i]; s += v; q += v * v; }
    ss[t] = s; sq[t] = q; __syncthreads();
    for (int o = 128; o > 0; o >>= 1) {
        if (t < o) { ss[t] += ss[t + o]; sq[t] += sq[t + o]; }
        __syncthreads();
    }
    if (t == 0) { p_sum[blockIdx.x] = ss[0]; p_sq[blockIdx.x] = sq[0]; }
}

__global__ __launch_bounds__(256) void k_reduce2(const float* __restrict__ p_sum,
                                                 const float* __restrict__ p_sq,
                                                 float* __restrict__ stats) {
    __shared__ float ss[256], sq[256];
    const int t = threadIdx.x;
    float s = 0.f, q = 0.f;
    for (int i = t; i < 2048; i += 256) { s += p_sum[i]; q += p_sq[i]; }
    ss[t] = s; sq[t] = q; __syncthreads();
    for (int o = 128; o > 0; o >>= 1) {
        if (t < o) { ss[t] += ss[t + o]; sq[t] += sq[t + o]; }
        __syncthreads();
    }
    if (t == 0) {
        const float N = (float)((size_t)B_N * S_IN);
        const float mean = ss[0] / N;
        const float var  = (sq[0] - N * mean * mean) / (N - 1.0f);  // ddof=1
        stats[0] = mean;
        stats[1] = rsqrtf(fmaxf(var, 1e-30f));
    }
}

// ------- stage 2: pack KxN f32 weights into bf16 WMMA panels ---------------
// dst[((k/32)*N + col)*32 + (k%32)]  -> a column's 32 K-values are 64B contiguous; a lane reads two 16B runs.
__global__ __launch_bounds__(256) void k_pack_panels(const float* __restrict__ src,
                                                     __bf16* __restrict__ dst,
                                                     int K, int N) {
    const int total = (K >> 5) * N;
    const int stride = gridDim.x * blockDim.x;
    for (int idx = blockIdx.x * blockDim.x + threadIdx.x; idx < total; idx += stride) {
        const int kb  = idx / N;
        const int col = idx - kb * N;
        const float* s = src + (size_t)(kb << 5) * N + col;
        __bf16* d = dst + (size_t)idx * 32;
#pragma unroll
        for (int kk = 0; kk < 32; ++kk) d[kk] = (__bf16)s[(size_t)kk * N];
    }
}

// ---------------- stage 3: H = tanh(xn @ w1 + b1), stored bf16 -------------
__global__ __launch_bounds__(128) void k_gemm1(const float* __restrict__ xp,
                                               const __bf16* __restrict__ w1t,
                                               const float* __restrict__ b1,
                                               const float* __restrict__ stats,
                                               __bf16* __restrict__ Hb) {
    const int wave = threadIdx.x >> 5;
    const int lane = threadIdx.x & 31;
    const int l16  = lane & 15;
    const int half = lane >> 4;
    const int tile = blockIdx.x * 4 + wave;      // 0..2047
    const int i0 = (tile >> 5) * 16;             // 64 M-tiles
    const int n0 = (tile & 31) * 16;             // 32 N-tiles
    const float mean = stats[0], rstd = stats[1];
    const float* aptr = xp + (size_t)(i0 + l16) * S_IN;
    const int kb = half * 8;
    v8f acc = {};
    for (int k = 0; k < S_IN; k += 32) {
        const float4* a0 = (const float4*)(aptr + k + kb);       // K kb..kb+7
        const float4* a1 = (const float4*)(aptr + k + kb + 16);  // K kb+16..kb+23
        float af[16];
        *(float4*)(af + 0)  = a0[0]; *(float4*)(af + 4)  = a0[1];
        *(float4*)(af + 8)  = a1[0]; *(float4*)(af + 12) = a1[1];
        v16bf a;
#pragma unroll
        for (int u = 0; u < 16; ++u) a[u] = (__bf16)((af[u] - mean) * rstd);
        const uint4* bq = (const uint4*)(w1t +
            (((size_t)(k >> 5) * HID + n0 + l16) << 5) + half * 8);
        V16 b; b.q[0] = bq[0]; b.q[1] = bq[2];
        acc = __builtin_amdgcn_wmma_f32_16x16x32_bf16(false, a, false, b.v,
                                                      (short)0, acc, false, false);
    }
    const float bias = b1[n0 + l16];
#pragma unroll
    for (int r = 0; r < 8; ++r) {
        const int row = i0 + half * 8 + r;
        Hb[(size_t)row * HID + n0 + l16] = (__bf16)fast_tanh(acc[r] + bias);
    }
}

// ---------------- stage 4: fused GEMM2 + circular spline -------------------
// Block = 4 waves. Each wave: 32 rows x 48 cols (= 2 spline sites x 24 ch for
// 32 batch rows): 6 WMMA accumulators over K=512 reusing one B-fragment set,
// epilogue through LDS, 2 splines per lane.
__global__ __launch_bounds__(128) void k_gemm2_spline(const __bf16* __restrict__ Hb,
                                                      const __bf16* __restrict__ w2t,
                                                      const float* __restrict__ b2,
                                                      const float* __restrict__ x_in,
                                                      const float* __restrict__ pshift,
                                                      float* __restrict__ phi_out,
                                                      float* __restrict__ ldp) {
    __shared__ float T[4][32][50];               // 32 rows x 48 cols (+2 pad) per wave
    const int wave = threadIdx.x >> 5;
    const int lane = threadIdx.x & 31;
    const int l16  = lane & 15;
    const int half = lane >> 4;
    const int i0   = blockIdx.y * 32;            // batch-row tile (32 rows)
    const int pair = blockIdx.x * 4 + wave;      // 0..2047 (s-pair)
    const int s0   = pair * 2;
    const size_t colbase = (size_t)s0 * NCH;     // 48-wide window into w2
    const __bf16* aptr0 = Hb + (size_t)(i0 + l16) * HID;
    const __bf16* aptr1 = aptr0 + (size_t)16 * HID;
    const int kb = half * 8;
    v8f acc0a = {}, acc1a = {}, acc2a = {};
    v8f acc0b = {}, acc1b = {}, acc2b = {};
#pragma unroll 2
    for (int k = 0; k < HID; k += 32) {
        V16 a0, a1;
        a0.q[0] = *(const uint4*)(aptr0 + k + kb);
        a0.q[1] = *(const uint4*)(aptr0 + k + kb + 16);
        a1.q[0] = *(const uint4*)(aptr1 + k + kb);
        a1.q[1] = *(const uint4*)(aptr1 + k + kb + 16);
        const __bf16* bp = w2t + (((size_t)(k >> 5) * W2COLS + colbase + l16) << 5)
                               + half * 8;
        if (k + 32 < HID) __builtin_prefetch(bp + ((size_t)W2COLS << 5), 0, 1);
        V16 b0, b1v, b2v;
        b0.q[0]  = *(const uint4*)(bp);            b0.q[1]  = *(const uint4*)(bp + 16);
        b1v.q[0] = *(const uint4*)(bp + 16 * 32);  b1v.q[1] = *(const uint4*)(bp + 16 * 32 + 16);
        b2v.q[0] = *(const uint4*)(bp + 32 * 32);  b2v.q[1] = *(const uint4*)(bp + 32 * 32 + 16);
        acc0a = __builtin_amdgcn_wmma_f32_16x16x32_bf16(false, a0.v, false, b0.v,
                                                        (short)0, acc0a, false, false);
        acc1a = __builtin_amdgcn_wmma_f32_16x16x32_bf16(false, a0.v, false, b1v.v,
                                                        (short)0, acc1a, false, false);
        acc2a = __builtin_amdgcn_wmma_f32_16x16x32_bf16(false, a0.v, false, b2v.v,
                                                        (short)0, acc2a, false, false);
        acc0b = __builtin_amdgcn_wmma_f32_16x16x32_bf16(false, a1.v, false, b0.v,
                                                        (short)0, acc0b, false, false);
        acc1b = __builtin_amdgcn_wmma_f32_16x16x32_bf16(false, a1.v, false, b1v.v,
                                                        (short)0, acc1b, false, false);
        acc2b = __builtin_amdgcn_wmma_f32_16x16x32_bf16(false, a1.v, false, b2v.v,
                                                        (short)0, acc2b, false, false);
    }
    // epilogue: net = tanh(acc + b2) into LDS (C/D layout: M = r + 8*half, N = l16)
    const float b0s = b2[colbase + l16];
    const float b1s = b2[colbase + 16 + l16];
    const float b2s = b2[colbase + 32 + l16];
#pragma unroll
    for (int r = 0; r < 8; ++r) {
        const int m = half * 8 + r;
        T[wave][m][l16]           = fast_tanh(acc0a[r] + b0s);
        T[wave][m][16 + l16]      = fast_tanh(acc1a[r] + b1s);
        T[wave][m][32 + l16]      = fast_tanh(acc2a[r] + b2s);
        T[wave][16 + m][l16]      = fast_tanh(acc0b[r] + b0s);
        T[wave][16 + m][16 + l16] = fast_tanh(acc1b[r] + b1s);
        T[wave][16 + m][32 + l16] = fast_tanh(acc2b[r] + b2s);
    }
    __syncthreads();

    // spline: lane -> (row m = rb*16 + l16, site sp = half); 24 channels from LDS
    const float shift = pshift[0];
#pragma unroll
    for (int rb = 0; rb < 2; ++rb) {
        const int m = rb * 16 + l16, sp = half;
        const int s = s0 + sp;
        const float* v = &T[wave][m][sp * NCH];
        float hr[8], wr[8], dr[8];
#pragma unroll
        for (int j = 0; j < 8; ++j) { hr[j] = v[j]; wr[j] = v[8 + j]; dr[j] = v[16 + j]; }
        float hm = hr[0], wm = wr[0];
#pragma unroll
        for (int j = 1; j < 8; ++j) { hm = fmaxf(hm, hr[j]); wm = fmaxf(wm, wr[j]); }
        float hn[8], wn[8], dn[8];
        float hs = 0.f, wsm = 0.f;
#pragma unroll
        for (int j = 0; j < 8; ++j) {
            hn[j] = __expf(hr[j] - hm); hs += hn[j];
            wn[j] = __expf(wr[j] - wm); wsm += wn[j];
            dn[j] = __logf(1.0f + __expf(dr[j]));    // softplus (|dr|<1, safe)
        }
        const float hsc = TWO_PI_F * fast_rcp(hs);
        const float wsc = TWO_PI_F * fast_rcp(wsm);
#pragma unroll
        for (int j = 0; j < 8; ++j) { hn[j] *= hsc; wn[j] *= wsc; }

        const float x = x_in[(size_t)(i0 + m) * S_OUT + s];
        // monotone bin search over cumsum boundaries c1..c7 (last true wins);
        // matches k = count(x_kp < x) - 1 clipped to [0,7]
        float cw = 0.f, ch = 0.f;
        float x_km1 = -EPS_F, phi_km1 = 0.f;
        float w_k = wn[0], h_k = hn[0], d_k = dn[0], d_kp1 = dn[1];
#pragma unroll
        for (int j = 0; j < 7; ++j) {
            cw += wn[j]; ch += hn[j];
            if (cw < x) {
                x_km1 = cw; phi_km1 = ch;
                w_k = wn[j + 1]; h_k = hn[j + 1];
                d_k = dn[j + 1]; d_kp1 = dn[(j + 2) & 7];
            }
        }
        const float inv_wk = fast_rcp(w_k);
        const float s_k    = h_k * inv_wk;
        const float alpha  = (x - x_km1) * inv_wk;
        const float a1m    = alpha * (1.f - alpha);
        const float denom  = s_k + (d_kp1 + d_k - 2.f * s_k) * a1m;
        const float inv_dn = fast_rcp(denom);
        float phi = phi_km1 + h_k * (s_k * alpha * alpha + d_k * a1m) * inv_dn;
        phi += shift;
        phi -= TWO_PI_F * floorf(phi * (1.f / TWO_PI_F));
        const float om   = 1.f - alpha;
        const float grad = s_k * s_k
                         * (d_kp1 * alpha * alpha + 2.f * s_k * a1m + d_k * om * om)
                         * inv_dn * inv_dn;
        const float lg = __logf(grad);
        phi_out[(size_t)(i0 + m) * S_OUT + s] = phi;
        const float lg2 = lg + __shfl_down(lg, 16, 32);   // combine sp=0/1 (wave32)
        if (half == 0) ldp[(size_t)(i0 + m) * NPAIRS + pair] = lg2;
    }
}

// ---------------- stage 5: deterministic log-density finalize --------------
__global__ __launch_bounds__(256) void k_ld_final(const float* __restrict__ ldp,
                                                  const float* __restrict__ ld_in,
                                                  float* __restrict__ out_ld) {
    __shared__ float ss[256];
    const int b = blockIdx.x, t = threadIdx.x;
    float s = 0.f;
    for (int i = t; i < NPAIRS; i += 256) s += ldp[(size_t)b * NPAIRS + i];
    ss[t] = s; __syncthreads();
    for (int o = 128; o > 0; o >>= 1) {
        if (t < o) ss[t] += ss[t + o];
        __syncthreads();
    }
    if (t == 0) out_ld[b] = ld_in[b] - ss[0];
}

extern "C" void kernel_launch(void* const* d_in, const int* in_sizes, int n_in,
                              void* d_out, int out_size, void* d_ws, size_t ws_size,
                              hipStream_t stream) {
    const float* x_in   = (const float*)d_in[0];
    const float* x_pass = (const float*)d_in[1];
    const float* ld_in  = (const float*)d_in[2];
    const float* w1     = (const float*)d_in[3];
    const float* b1     = (const float*)d_in[4];
    const float* w2     = (const float*)d_in[5];
    const float* b2     = (const float*)d_in[6];
    const float* pshift = (const float*)d_in[7];

    float* ws    = (float*)d_ws;
    float* p_sum = ws;                                   // 2048
    float* p_sq  = ws + 2048;                            // 2048
    float* stats = ws + 4096;                            // 64 (padded)
    float* ldp   = ws + 4160;                            // 1024*2048
    __bf16* w1t  = (__bf16*)(ldp + (size_t)B_N * NPAIRS);// 4096*512 bf16 panels
    __bf16* Hb   = w1t + (size_t)S_IN * HID;             // 1024*512 bf16
    __bf16* w2t  = Hb + (size_t)B_N * HID;               // 512*98304 bf16 panels

    float* phi_out = (float*)d_out;
    float* out_ld  = phi_out + (size_t)B_N * S_OUT;

    k_reduce1<<<2048, 256, 0, stream>>>(x_pass, p_sum, p_sq);
    k_reduce2<<<1, 256, 0, stream>>>(p_sum, p_sq, stats);
    k_pack_panels<<<256, 256, 0, stream>>>(w1, w1t, S_IN, HID);
    k_pack_panels<<<4096, 256, 0, stream>>>(w2, w2t, HID, W2COLS);
    k_gemm1<<<512, 128, 0, stream>>>(x_pass, w1t, b1, stats, Hb);
    dim3 g2(512, 32);
    k_gemm2_spline<<<g2, 128, 0, stream>>>(Hb, w2t, b2, x_in, pshift, phi_out, ldp);
    k_ld_final<<<1024, 256, 0, stream>>>(ldp, ld_in, out_ld);
}
